// HeteroEncoder_40939628265668
// MI455X (gfx1250) — matrix operand order repaired and checked by reading, not compiled, not run on hardware
//
#include <hip/hip_runtime.h>
#include <stdint.h>

typedef _Float16 v16h __attribute__((ext_vector_type(16)));
typedef _Float16 v8h  __attribute__((ext_vector_type(8)));
typedef float    v8f  __attribute__((ext_vector_type(8)));

#define HIDDEN 256
#define K1 64          // combined layer-1 K (32 var + 16 clause + 2 bias + pad)
#define K2 512         // combined layer-2 K (256 var + 256 clause)
#define TM  128        // rows per block
#define NEG_SLOPE 0.01f
#define F_PAD 8
#define H_PAD 8

// ---------------------------------------------------------------------------
// Prep: pack combined weight matrices into workspace as f16, B-transposed
// layout (N-major, K contiguous) so a WMMA B-fragment is a contiguous read.
//   W1cT[n][k], k<32 -> vW1[k][n]; 32<=k<48 -> cW1[k-32][n];
//               k==48 -> vb1[n];  k==49 -> cb1[n]; else 0
//   W2cT[n][k], k<256 -> vW2[k][n]; else cW2[k-256][n]
// ---------------------------------------------------------------------------
__global__ __launch_bounds__(256) void prep_weights(
    const float* __restrict__ vW1, const float* __restrict__ vb1,
    const float* __restrict__ cW1, const float* __restrict__ cb1,
    const float* __restrict__ vW2, const float* __restrict__ cW2,
    _Float16* __restrict__ W1cT, _Float16* __restrict__ W2cT)
{
    int idx = blockIdx.x * 256 + threadIdx.x;
    if (idx < HIDDEN * K1) {
        int n = idx >> 6, k = idx & 63;
        float v = 0.0f;
        if (k < 32)       v = vW1[k * HIDDEN + n];
        else if (k < 48)  v = cW1[(k - 32) * HIDDEN + n];
        else if (k == 48) v = vb1[n];
        else if (k == 49) v = cb1[n];
        W1cT[n * K1 + k] = (_Float16)v;
    } else {
        int j = idx - HIDDEN * K1;
        if (j < HIDDEN * K2) {
            int n = j >> 9, k = j & 511;
            float v = (k < 256) ? vW2[k * HIDDEN + n]
                                : cW2[(k - 256) * HIDDEN + n];
            W2cT[(size_t)n * K2 + k] = (_Float16)v;
        }
    }
}

// Async global->LDS copy of one 32-byte fragment slot (2 x b128), ASYNCcnt.
// LDS address = low 32 bits of the flat shared pointer (aperture maps 1:1).
__device__ __forceinline__ void async_copy_slot(const _Float16* gsrc, _Float16* ldst)
{
    unsigned ldsa = (unsigned)(uintptr_t)ldst;
    asm volatile("global_load_async_to_lds_b128 %0, %1, off"
                 :: "v"(ldsa), "v"(gsrc) : "memory");
    asm volatile("global_load_async_to_lds_b128 %0, %1, off offset:32"
                 :: "v"(ldsa), "v"(gsrc) : "memory");
}

// ---------------------------------------------------------------------------
// Main kernel: 8 waves x 16 rows = 128 rows per block, 256 output cols.
// Weights are staged into LDS in "fragment order": slot s = ((ct*KB+kb)*32+lane)
// holds the 32 bytes lane `lane` needs for B-fragment (ct,kb), so the phase
// B-load is one contiguous conflict-free ds_load pair.
// ---------------------------------------------------------------------------
__global__ __launch_bounds__(256) void hetero_encoder(
    const float* __restrict__ x,
    const _Float16* __restrict__ W1cT,
    const _Float16* __restrict__ W2cT,
    const float* __restrict__ vb2,
    const float* __restrict__ cb2,
    float* __restrict__ out,
    int nNodes)
{
    __shared__ __align__(16) _Float16 Fs[TM][K1 + F_PAD];          // 18.4 KB
    __shared__ __align__(16) _Float16 Hs[8][16][HIDDEN + H_PAD];   // 67.6 KB
    __shared__ __align__(16) _Float16 Ws1[16 * 2 * 32 * 16];       // 32 KB, frag order
    __shared__ __align__(16) _Float16 Bs[2 * 16 * 32 * 16];        // 32 KB, frag order
    __shared__ float tpy[TM];                                       // total ~148.5 KB

    const int tid  = threadIdx.x;
    const int wave = tid >> 5;
    const int lane = tid & 31;
    const int m    = lane & 15;     // A-row (and B-col) owned by this lane
    const int hsel = lane >> 4;     // lane half selects K sub-range
    const long rowBase = (long)blockIdx.x * TM;

    // ---- Phase 0a: kick off async staging of all W1c fragments ----
    // 16 ct-tiles x 2 kb x 32 lanes = 1024 slots; 4 per thread.
#pragma unroll
    for (int i = 0; i < 4; ++i) {
        int s     = tid + i * 256;
        int laneS = s & 31;
        int kbS   = (s >> 5) & 1;
        int ctS   = s >> 6;
        int nS    = ctS * 16 + (laneS & 15);
        async_copy_slot(W1cT + nS * K1 + kbS * 32 + (laneS >> 4) * 8,
                        &Ws1[s * 16]);
    }

    // ---- Phase 0b: node types + combined layer-1 feature tile in LDS ----
    if (tid < TM) {
        long r = rowBase + tid;
        tpy[tid] = x[(r < nNodes ? r : (long)nNodes - 1) * 33];
    }
    __syncthreads();

    for (int e = tid; e < TM * K1; e += 256) {
        int r = e >> 6, k = e & 63;
        long gr = rowBase + r;
        if (gr >= nNodes) gr = nNodes - 1;
        bool isVar = (tpy[r] == 0.0f);
        float v = 0.0f;
        if (k < 32)       v = isVar ? x[gr * 33 + 1 + k] : 0.0f;
        else if (k < 48)  v = isVar ? 0.0f : x[gr * 33 + 1 + (k - 32)];
        else if (k == 48) v = isVar ? 1.0f : 0.0f;
        else if (k == 49) v = isVar ? 0.0f : 1.0f;
        Fs[r][k] = (_Float16)v;
    }
    asm volatile("s_wait_asynccnt 0x0" ::: "memory");   // W1 staging done (this wave)
    __syncthreads();                                    // Fs + Ws1 visible block-wide

    const int r0   = wave * 16;     // this wave's 16 rows
    const int koff = hsel * 8;      // K sub-offset within a 32-wide chunk

    // ---- Phase 1: h = leakyrelu(F @ W1c), per-wave 16x256 tile ----
    // A-fragment layout (16-bit A 16x32): lane holds row M = lane%16;
    // halves 0..7 = K[kb*32+koff .. +7], halves 8..15 = K[kb*32+16+koff .. +7]
    v16h a1[2];
#pragma unroll
    for (int kb = 0; kb < 2; ++kb) {
        v8h lo = *(const v8h*)&Fs[r0 + m][kb * 32 + koff];
        v8h hi = *(const v8h*)&Fs[r0 + m][kb * 32 + 16 + koff];
        v16h a;
#pragma unroll
        for (int i = 0; i < 8; ++i) { a[i] = lo[i]; a[i + 8] = hi[i]; }
        a1[kb] = a;
    }

    for (int ct = 0; ct < 16; ++ct) {
        int n = ct * 16 + m;
        v8f c = {};
#pragma unroll
        for (int kb = 0; kb < 2; ++kb) {
            v16h b = *(const v16h*)&Ws1[((ct * 2 + kb) * 32 + lane) * 16];
            c = __builtin_amdgcn_wmma_f32_16x16x32_f16(
                    false, a1[kb], false, b, (short)0, c, false, false);
        }
        // C layout: vgpr v -> row M = v + hsel*8, col N = n
#pragma unroll
        for (int v = 0; v < 8; ++v) {
            float h = c[v];
            h = (h >= 0.0f) ? h : NEG_SLOPE * h;
            Hs[wave][v + hsel * 8][n] = (_Float16)h;
        }
    }
    // Hs tile is wave-private; same-wave LDS ops are in order -> no barrier.

    // ---- Phase 2: out = [h|0 , 0|h] @ [vW2 ; cW2] + b2(type) ----
    v16h Afr[8];
#pragma unroll
    for (int q = 0; q < 8; ++q) {
        v8h lo = *(const v8h*)&Hs[wave][m][q * 32 + koff];
        v8h hi = *(const v8h*)&Hs[wave][m][q * 32 + 16 + koff];
        v16h a;
#pragma unroll
        for (int i = 0; i < 8; ++i) { a[i] = lo[i]; a[i + 8] = hi[i]; }
        Afr[q] = a;
    }

    const v16h zf = {};
    const bool isVarRow = (tpy[r0 + m] == 0.0f);   // type of the A-row this lane owns
    float t8[8];
#pragma unroll
    for (int v = 0; v < 8; ++v) t8[v] = tpy[r0 + v + hsel * 8];

    // 8 column-chunks of 32 cols; W2c chunk staged to LDS once, reused by 8 waves.
    for (int cc = 0; cc < 8; ++cc) {
        __syncthreads();            // everyone done reading previous Bs chunk
        // stage 2 ct-tiles x 16 kb x 32 lanes = 1024 slots, 4 per thread (async)
#pragma unroll
        for (int i = 0; i < 4; ++i) {
            int s     = tid + i * 256;
            int laneS = s & 31;
            int kbS   = (s >> 5) & 15;
            int ctS   = s >> 9;
            int nS    = cc * 32 + ctS * 16 + (laneS & 15);
            async_copy_slot(W2cT + (size_t)nS * K2 + kbS * 32 + (laneS >> 4) * 8,
                            &Bs[s * 16]);
        }
        asm volatile("s_wait_asynccnt 0x0" ::: "memory");
        __syncthreads();            // chunk visible block-wide

#pragma unroll
        for (int ct2 = 0; ct2 < 2; ++ct2) {
            int n = cc * 32 + ct2 * 16 + m;
            v8f c = {};
#pragma unroll
            for (int kb = 0; kb < 16; ++kb) {
                // kb<8 is the vW2 K-half; a var row contributes only there,
                // a clause row only to kb>=8. Per-lane select (row lives in lane).
                bool act = ((kb < 8) == isVarRow);
                v16h a = act ? Afr[kb & 7] : zf;
                v16h b = *(const v16h*)&Bs[((ct2 * 16 + kb) * 32 + lane) * 16];
                c = __builtin_amdgcn_wmma_f32_16x16x32_f16(
                        false, a, false, b, (short)0, c, false, false);
            }
            float bv = vb2[n], bc = cb2[n];
#pragma unroll
            for (int v = 0; v < 8; ++v) {
                long gr = rowBase + r0 + v + hsel * 8;
                if (gr < nNodes) {
                    float o = c[v] + ((t8[v] == 0.0f) ? bv : bc);
                    __builtin_nontemporal_store(o, &out[gr * (long)HIDDEN + n]);
                }
            }
        }
    }
}

// ---------------------------------------------------------------------------
extern "C" void kernel_launch(void* const* d_in, const int* in_sizes, int n_in,
                              void* d_out, int out_size, void* d_ws, size_t ws_size,
                              hipStream_t stream) {
    const float* x   = (const float*)d_in[0];
    const float* vW1 = (const float*)d_in[1];
    const float* vb1 = (const float*)d_in[2];
    const float* vW2 = (const float*)d_in[3];
    const float* vb2 = (const float*)d_in[4];
    const float* cW1 = (const float*)d_in[5];
    const float* cb1 = (const float*)d_in[6];
    const float* cW2 = (const float*)d_in[7];
    const float* cb2 = (const float*)d_in[8];
    float* out = (float*)d_out;

    int nNodes = in_sizes[0] / 33;

    _Float16* W1cT = (_Float16*)d_ws;           // 256*64  f16 = 32 KB
    _Float16* W2cT = W1cT + HIDDEN * K1;        // 256*512 f16 = 256 KB

    int prepElems = HIDDEN * K1 + HIDDEN * K2;
    prep_weights<<<(prepElems + 255) / 256, 256, 0, stream>>>(
        vW1, vb1, cW1, cb1, vW2, cW2, W1cT, W2cT);

    int nTiles = (nNodes + TM - 1) / TM;        // 400000/128 = 3125 exactly
    hetero_encoder<<<nTiles, 256, 0, stream>>>(
        x, W1cT, W2cT, vb2, cb2, out, nNodes);
}
